// ResGCN_27367531610148
// MI455X (gfx1250) — matrix operand order repaired and checked by reading, not compiled, not run on hardware
//
#include <hip/hip_runtime.h>
#include <hip/hip_bf16.h>

#define DFEAT 64
#define BST_STRIDE 72   // bf16 W tile: 144B row stride -> conflict-free reads
#define ASF_STRIDE 68   // fp32 A tile: TDM pads 4 dwords per 64 -> 272B stride

typedef __attribute__((ext_vector_type(16))) __bf16 v16bf;
typedef __attribute__((ext_vector_type(8)))  __bf16 v8bf;
typedef __attribute__((ext_vector_type(8)))  float  v8f;
typedef __attribute__((ext_vector_type(4)))  unsigned int uint32x4;
typedef __attribute__((ext_vector_type(8)))  int  int32x8;
typedef __attribute__((ext_vector_type(4)))  int  int32x4;

// ---------------------------------------------------------------------------
// Generic  Out[M,64] = A[M,64] @ W[64,64] (+ bias)  via v_wmma_f32_16x16x32_bf16
// Block = 128 threads (4 waves), one 64x64 output tile per block.
// A tile DMA'd global->LDS by the Tensor Data Mover (async, M-tail clipped to
// zero by tensor_dim1, 4-dword pad per 64 dwords for bank-conflict-free reads).
// W staged transposed (column-major bf16) so B fragments are contiguous in K.
// ---------------------------------------------------------------------------
__global__ __launch_bounds__(128)
void gemm64_wmma_bf16(const float* __restrict__ A,
                      const float* __restrict__ W,
                      const float* __restrict__ bias,
                      float* __restrict__ Out, int M)
{
    __shared__ float  AsF[64 * ASF_STRIDE];    // fp32 A tile (TDM destination)
    __shared__ __bf16 Bst[64 * BST_STRIDE];    // W transposed, bf16

    const int tid  = threadIdx.x;
    const int wave = tid >> 5;
    const int lane = tid & 31;
    const int r0   = blockIdx.x * 64;

    // ---- wave 0: issue TDM load of the A tile (rows r0..r0+63) ----
    if (wave == 0) {
        unsigned lds_addr   = (unsigned)(unsigned long long)(void*)&AsF[0];
        unsigned long long ga = (unsigned long long)A + (unsigned long long)r0 * (DFEAT * 4);
        unsigned rows = (unsigned)(M - r0);            // clip: OOB rows read as 0

        uint32x4 g0;
        g0.x = 1u;                                     // count=1, no gather
        g0.y = lds_addr;                               // lds_addr [63:32]
        g0.z = (unsigned)ga;                           // global_addr[31:0]
        g0.w = (unsigned)((ga >> 32) & 0x1FFFFFFu) | (2u << 30);  // addr[56:32] | type=2

        int32x8 g1;
        g1[0] = (int)((2u << 16)        // data_size = 4B
                    | (1u << 20)        // pad_enable
                    | (5u << 22)        // pad_interval: 2^5*8B = 64 dwords
                    | (3u << 25));      // pad_amount: 4 dwords
        g1[1] = (int)(64u << 16);       // tensor_dim0[15:0] = 64
        g1[2] = (int)((rows & 0xFFFFu) << 16);                 // tensor_dim1[15:0]
        g1[3] = (int)(((rows >> 16) & 0xFFFFu) | (64u << 16)); // dim1 hi | tile_dim0=64
        g1[4] = 64;                     // tile_dim1 = 64 (tile_dim2 = 0)
        g1[5] = 64;                     // tensor_dim0_stride = 64
        g1[6] = 0;
        g1[7] = 0;

        int32x4 g2 = {0, 0, 0, 0};      // 2D tensor: groups 2/3 unused
        int32x4 g3 = {0, 0, 0, 0};
        int32x8 g4 = {0, 0, 0, 0, 0, 0, 0, 0};   // extra operand (6-arg form)

        __builtin_amdgcn_tensor_load_to_lds(g0, g1, g2, g3, g4, 0);
        __builtin_amdgcn_s_wait_tensorcnt(0);
    }

    // ---- all waves: stage W (64x64 fp32 -> bf16, transposed scatter) ----
    #pragma unroll
    for (int i = 0; i < 8; ++i) {
        int idx = (i * 128 + tid) * 4;
        int k   = idx >> 6;        // row of W  (input dim)
        int n0  = idx & 63;        // col of W  (output dim)
        float4 w4 = *reinterpret_cast<const float4*>(W + idx);
        Bst[(n0 + 0) * BST_STRIDE + k] = (__bf16)w4.x;
        Bst[(n0 + 1) * BST_STRIDE + k] = (__bf16)w4.y;
        Bst[(n0 + 2) * BST_STRIDE + k] = (__bf16)w4.z;
        Bst[(n0 + 3) * BST_STRIDE + k] = (__bf16)w4.w;
    }
    __syncthreads();

    const int ln = lane & 15;   // N col within tile / M row within wave tile
    const int lh = lane >> 4;   // lane-half selector

    // Accumulators initialized with bias (broadcast along M).
    v8f c[4];
    #pragma unroll
    for (int nt = 0; nt < 4; ++nt) {
        float bv = bias ? bias[nt * 16 + ln] : 0.0f;
        #pragma unroll
        for (int r = 0; r < 8; ++r) c[nt][r] = bv;
    }

    const float* arow = &AsF[(wave * 16 + ln) * ASF_STRIDE];

    #pragma unroll
    for (int k0 = 0; k0 < 64; k0 += 32) {
        // A fragment (16-bit A 16x32 layout): lane-half adds +8 to K;
        // elements 0..7 and 8..15 each contiguous -> 4x ds_load_b128 + cvt.
        float4 f0 = *reinterpret_cast<const float4*>(arow + k0 + 8 * lh);
        float4 f1 = *reinterpret_cast<const float4*>(arow + k0 + 8 * lh + 4);
        float4 f2 = *reinterpret_cast<const float4*>(arow + k0 + 8 * lh + 16);
        float4 f3 = *reinterpret_cast<const float4*>(arow + k0 + 8 * lh + 20);
        v16bf a = { (__bf16)f0.x, (__bf16)f0.y, (__bf16)f0.z, (__bf16)f0.w,
                    (__bf16)f1.x, (__bf16)f1.y, (__bf16)f1.z, (__bf16)f1.w,
                    (__bf16)f2.x, (__bf16)f2.y, (__bf16)f2.z, (__bf16)f2.w,
                    (__bf16)f3.x, (__bf16)f3.y, (__bf16)f3.z, (__bf16)f3.w };

        #pragma unroll
        for (int nt = 0; nt < 4; ++nt) {
            // B fragment: N = ln, K = k0 + 8*lh + e (e<8), k0 + 16 + 8*lh + (e-8) (e>=8).
            const __bf16* bcol = &Bst[(nt * 16 + ln) * BST_STRIDE + k0 + 8 * lh];
            v8bf b0 = *reinterpret_cast<const v8bf*>(bcol);
            v8bf b1 = *reinterpret_cast<const v8bf*>(bcol + 16);
            v16bf b = __builtin_shufflevector(b0, b1, 0,1,2,3,4,5,6,7,8,9,10,11,12,13,14,15);
            c[nt] = __builtin_amdgcn_wmma_f32_16x16x32_bf16(
                        false, a, false, b, (short)0, c[nt], false, false);
        }
    }

    // C/D layout: VGPR r -> M = r + 8*lh, N = ln.
    if (r0 + 64 <= M) {
        // fast path: full tile, unguarded stores (no exec-mask churn)
        #pragma unroll
        for (int nt = 0; nt < 4; ++nt) {
            #pragma unroll
            for (int r = 0; r < 8; ++r) {
                int row = r0 + wave * 16 + r + 8 * lh;
                Out[(size_t)row * 64 + nt * 16 + ln] = c[nt][r];
            }
        }
    } else {
        #pragma unroll
        for (int nt = 0; nt < 4; ++nt) {
            #pragma unroll
            for (int r = 0; r < 8; ++r) {
                int row = r0 + wave * 16 + r + 8 * lh;
                if (row < M) Out[(size_t)row * 64 + nt * 16 + ln] = c[nt][r];
            }
        }
    }
}

// ---------------------------------------------------------------------------
// Elementwise helpers
// ---------------------------------------------------------------------------
__global__ void relu_kernel(const float* __restrict__ in, float* __restrict__ out, size_t n)
{
    size_t i = (size_t)blockIdx.x * blockDim.x + threadIdx.x;
    if (i < n) out[i] = fmaxf(in[i], 0.0f);
}

__global__ void add_kernel(const float* __restrict__ a, const float* __restrict__ b,
                           float* __restrict__ out, size_t n)
{
    size_t i = (size_t)blockIdx.x * blockDim.x + threadIdx.x;
    if (i < n) out[i] = a[i] + b[i];
}

__global__ void init_softmax_state(float* __restrict__ m, float* __restrict__ denom, int N)
{
    int i = blockIdx.x * blockDim.x + threadIdx.x;
    if (i < N) { m[i] = -3.0e38f; denom[i] = 0.0f; }
}

// ---------------------------------------------------------------------------
// Edge pass 1: logits[e] = dot(q[dst], k[src]+e_feat)/8 ; segment max (atomic)
// ---------------------------------------------------------------------------
__global__ __launch_bounds__(256)
void edge_logits_kernel(const int* __restrict__ ei,
                        const float* __restrict__ q,
                        const float* __restrict__ kbuf,
                        const float* __restrict__ ebuf,
                        float* __restrict__ logits,
                        float* __restrict__ mbuf, int E)
{
    int e = blockIdx.x * blockDim.x + threadIdx.x;
    if (e >= E) return;
    int src = ei[e];
    int dst = ei[E + e];
    const float4* qp = reinterpret_cast<const float4*>(q    + (size_t)dst * 64);
    const float4* kp = reinterpret_cast<const float4*>(kbuf + (size_t)src * 64);
    const float4* ep = reinterpret_cast<const float4*>(ebuf + (size_t)e   * 64);
    float acc = 0.0f;
    #pragma unroll
    for (int i = 0; i < 16; ++i) {
        float4 qa = qp[i], ka = kp[i], ea = ep[i];
        acc += qa.x * (ka.x + ea.x) + qa.y * (ka.y + ea.y)
             + qa.z * (ka.z + ea.z) + qa.w * (ka.w + ea.w);
    }
    float lg = acc * 0.125f;               // 1/sqrt(64)
    logits[e] = lg;
    atomicMax(mbuf + dst, lg);             // global_atomic_max_num_f32
}

// ---------------------------------------------------------------------------
// Edge pass 2: a[e] = exp(logit - m[dst]); denom[dst] += a (atomic)
// ---------------------------------------------------------------------------
__global__ __launch_bounds__(256)
void edge_exp_kernel(const int* __restrict__ ei,
                     const float* __restrict__ logits,
                     const float* __restrict__ mbuf,
                     float* __restrict__ abuf,
                     float* __restrict__ denom, int E)
{
    int e = blockIdx.x * blockDim.x + threadIdx.x;
    if (e >= E) return;
    int dst = ei[E + e];
    float a = __expf(logits[e] - mbuf[dst]);
    abuf[e] = a;
    atomicAdd(denom + dst, a);             // global_atomic_add_f32
}

// ---------------------------------------------------------------------------
// Edge pass 3: acc[dst] += (v[src] + e_feat) * (a[e]/denom[dst])
// 16 lanes per edge, float4 per lane -> coalesced e-stream + coalesced atomics
// ---------------------------------------------------------------------------
__global__ __launch_bounds__(256)
void edge_scatter_kernel(const int* __restrict__ ei,
                         const float* __restrict__ vbuf,
                         const float* __restrict__ ebuf,
                         const float* __restrict__ abuf,
                         const float* __restrict__ denom,
                         float* __restrict__ acc, int E)
{
    size_t gid = (size_t)blockIdx.x * blockDim.x + threadIdx.x;
    if (gid >= (size_t)E * 16) return;
    int e  = (int)(gid >> 4);
    int c4 = (int)(gid & 15);
    int src = ei[e];
    int dst = ei[E + e];
    float alpha = abuf[e] / denom[dst];
    float4 vv = *reinterpret_cast<const float4*>(vbuf + (size_t)src * 64 + c4 * 4);
    float4 ee = *reinterpret_cast<const float4*>(ebuf + (size_t)e   * 64 + c4 * 4);
    float* dp = acc + (size_t)dst * 64 + c4 * 4;
    atomicAdd(dp + 0, (vv.x + ee.x) * alpha);
    atomicAdd(dp + 1, (vv.y + ee.y) * alpha);
    atomicAdd(dp + 2, (vv.z + ee.z) * alpha);
    atomicAdd(dp + 3, (vv.w + ee.w) * alpha);
}

// ---------------------------------------------------------------------------
// Host orchestration
// ---------------------------------------------------------------------------
static inline void run_gemm(const float* A, const float* W, const float* bias,
                            float* Out, int M, hipStream_t s)
{
    gemm64_wmma_bf16<<<(M + 63) / 64, 128, 0, s>>>(A, W, bias, Out, M);
}

extern "C" void kernel_launch(void* const* d_in, const int* in_sizes, int n_in,
                              void* d_out, int out_size, void* d_ws, size_t ws_size,
                              hipStream_t stream)
{
    const int N = in_sizes[0] / DFEAT;
    const int E = in_sizes[1] / 2;

    const float* x  = (const float*)d_in[0];
    const int*   ei = (const int*)  d_in[1];
    const float* ea = (const float*)d_in[2];
    const float* Wq1 = (const float*)d_in[3],  *Wk1 = (const float*)d_in[4];
    const float* Wv1 = (const float*)d_in[5],  *We1 = (const float*)d_in[6];
    const float* Ws1 = (const float*)d_in[7];
    const float* Wq2 = (const float*)d_in[8],  *Wk2 = (const float*)d_in[9];
    const float* Wv2 = (const float*)d_in[10], *We2 = (const float*)d_in[11];
    const float* Ws2 = (const float*)d_in[12];
    const float* bq1 = (const float*)d_in[13], *bk1 = (const float*)d_in[14];
    const float* bv1 = (const float*)d_in[15], *bs1 = (const float*)d_in[16];
    const float* bq2 = (const float*)d_in[17], *bk2 = (const float*)d_in[18];
    const float* bv2 = (const float*)d_in[19], *bs2 = (const float*)d_in[20];

    // Workspace layout (all float4-aligned)
    float* ws   = (float*)d_ws;
    size_t nf   = (size_t)N * DFEAT;
    float* xr     = ws;                       // N*64  relu input
    float* qb     = xr   + nf;                // N*64
    float* kb     = qb   + nf;                // N*64
    float* vb     = kb   + nf;                // N*64
    float* acc    = vb   + nf;                // N*64  skip + aggregated messages
    float* ebuf   = acc  + nf;                // E*64  edge features after We
    float* logits = ebuf + (size_t)E * DFEAT; // E
    float* abuf   = logits + E;               // E
    float* mbuf   = abuf   + E;               // N
    float* denom  = mbuf   + N;               // N

    const int T = 256;
    const int gN  = (int)((nf + T - 1) / T);
    const int gE  = (E + T - 1) / T;
    const int gE16 = (int)(((size_t)E * 16 + T - 1) / T);
    const int gNn = (N + T - 1) / T;

    const float* Wq[2] = {Wq1, Wq2}; const float* Wk[2] = {Wk1, Wk2};
    const float* Wv[2] = {Wv1, Wv2}; const float* We[2] = {We1, We2};
    const float* Wsk[2] = {Ws1, Ws2};
    const float* bq[2] = {bq1, bq2}; const float* bk[2] = {bk1, bk2};
    const float* bv[2] = {bv1, bv2}; const float* bs[2] = {bs1, bs2};

    for (int L = 0; L < 2; ++L) {
        const float* h_in = (L == 0) ? x : acc;           // layer input
        relu_kernel<<<gN, T, 0, stream>>>(h_in, xr, nf);

        run_gemm(xr, Wq[L],  bq[L], qb,   N, stream);
        run_gemm(xr, Wk[L],  bk[L], kb,   N, stream);
        run_gemm(xr, Wv[L],  bv[L], vb,   N, stream);
        run_gemm(ea, We[L],  nullptr, ebuf, E, stream);
        run_gemm(xr, Wsk[L], bs[L], acc,  N, stream);     // acc = skip term

        init_softmax_state<<<gNn, T, 0, stream>>>(mbuf, denom, N);
        edge_logits_kernel<<<gE, T, 0, stream>>>(ei, qb, kb, ebuf, logits, mbuf, E);
        edge_exp_kernel<<<gE, T, 0, stream>>>(ei, logits, mbuf, abuf, denom, E);
        edge_scatter_kernel<<<gE16, T, 0, stream>>>(ei, vb, ebuf, abuf, denom, acc, E);
    }

    // out = x + layer2 output (identity residual)
    add_kernel<<<gN, T, 0, stream>>>(x, acc, (float*)d_out, nf);
}
